// StaticSelfAttention_77412490543627
// MI455X (gfx1250) — compile-verified, matrix operand order checked by reading, not run on hardware
//
#include <hip/hip_runtime.h>
#include <hip/hip_bf16.h>

typedef __bf16 bf16;
typedef __attribute__((ext_vector_type(16))) __bf16 bf16x16;
typedef __attribute__((ext_vector_type(8)))  __bf16 bf16x8;
typedef __attribute__((ext_vector_type(8)))  float  f32x8;

static constexpr int S_   = 2048;
static constexpr int DIM_ = 5120;
static constexpr int H_   = 40;
static constexpr int D_   = 128;
static constexpr int N3_  = 3 * DIM_;          // 15360
static constexpr float SCALE_ = 0.08838834764831845f;  // D^-0.5

// Low 32 bits of a generic LDS pointer are the workgroup-relative LDS byte
// address (ISA 10.2: LDS aperture addresses truncate to LDS_ADDR.U32).
#define LDS_OFF(p) ((unsigned)(unsigned long long)(p))

// ---------------------------------------------------------------------------
// CDNA5 async global->LDS copy (ASYNCcnt-tracked) + waits
// ---------------------------------------------------------------------------
__device__ __forceinline__ void async_copy_b128(unsigned lds_byte_off, const void* gptr) {
  asm volatile("global_load_async_to_lds_b128 %0, %1, off"
               :: "v"(lds_byte_off), "v"(gptr)
               : "memory");
}
__device__ __forceinline__ void wait_async_le4() {
  asm volatile("s_wait_asynccnt 0x4" ::: "memory");
}
__device__ __forceinline__ void wait_async_0() {
  asm volatile("s_wait_asynccnt 0x0" ::: "memory");
}

// ---------------------------------------------------------------------------
// WMMA helpers (CDNA5 16x16x32 bf16, fp32 accumulate)
// ---------------------------------------------------------------------------
__device__ __forceinline__ f32x8 wmma_bf16(bf16x16 a, bf16x16 b, f32x8 c) {
  return __builtin_amdgcn_wmma_f32_16x16x32_bf16(
      /*neg_a=*/false, a, /*neg_b=*/false, b,
      /*c_mod=*/(short)0, c, /*reuse_a=*/false, /*reuse_b=*/false);
}

// Load one 16x32 bf16 operand fragment (A, or B supplied K-contiguous, i.e.
// [16 rows][K] with leading dimension ld). Per-lane K mapping per ISA:
// lanes 0-15: K {0..7, 16..23}; lanes 16-31: K {8..15, 24..31}.
__device__ __forceinline__ bf16x16 load_frag16(const bf16* base, int ld, int lane) {
  const int r = lane & 15;
  const int h = lane >> 4;
  const bf16* row = base + (size_t)r * ld + h * 8;
  union { bf16x16 v; bf16x8 p[2]; } u;
  u.p[0] = *(const bf16x8*)(row);
  u.p[1] = *(const bf16x8*)(row + 16);
  return u.v;
}

// ---------------------------------------------------------------------------
// fp32 -> bf16 elementwise convert
// ---------------------------------------------------------------------------
__global__ void convert_bf16(const float* __restrict__ src, bf16* __restrict__ dst, int n) {
  int i = blockIdx.x * blockDim.x + threadIdx.x;
  if (i < n) dst[i] = (bf16)src[i];
}

// ---------------------------------------------------------------------------
// Transpose + convert: src [R][C] fp32 -> dst [C][R] bf16  (R,C multiples of 32)
// ---------------------------------------------------------------------------
__global__ void transpose_convert(const float* __restrict__ src, bf16* __restrict__ dst,
                                  int R, int C) {
  __shared__ float tile[32][33];
  const int c0 = blockIdx.x * 32, r0 = blockIdx.y * 32;
  const int tx = threadIdx.x, ty = threadIdx.y;
#pragma unroll
  for (int i = 0; i < 4; ++i)
    tile[ty + i * 8][tx] = src[(size_t)(r0 + ty + i * 8) * C + (c0 + tx)];
  __syncthreads();
#pragma unroll
  for (int i = 0; i < 4; ++i)
    dst[(size_t)(c0 + ty + i * 8) * R + (r0 + tx)] = (bf16)tile[tx][ty + i * 8];
}

// ---------------------------------------------------------------------------
// GEMM: C[M][N] fp32 = A[M][K] bf16 @ Bt[N][K] bf16 + bias[N]
// Block tile 128x128, k-step 32, double-buffered async global->LDS staging.
// 8 waves in 2(M) x 4(N); each wave computes 64x32 (4x2 WMMA tiles).
// M, N multiples of 128; K multiple of 32.
// ---------------------------------------------------------------------------
__global__ __launch_bounds__(256) void gemm_bf16(const bf16* __restrict__ A,
                                                 const bf16* __restrict__ Bt,
                                                 const float* __restrict__ bias,
                                                 float* __restrict__ C,
                                                 int M, int N, int K) {
  __shared__ __align__(16) bf16 Asm[2][128 * 32];   // [buf][row*32 + k] 8KB/buf
  __shared__ __align__(16) bf16 Bsm[2][128 * 32];

  const int tid = threadIdx.x;
  const int lane = tid & 31, wave = tid >> 5;
  const int wm = wave & 1, wn = wave >> 1;          // 2 x 4 wave grid
  const int m0 = blockIdx.y * 128;
  const int n0 = blockIdx.x * 128;

  // Copy assignment: thread -> (row, 16B chunk); 2 rows each for A and B.
  const int crow  = tid >> 2;          // 0..63
  const int cbyte = (tid & 3) * 16;    // 0,16,32,48 within a 64B row

  const char* Ab0 = (const char*)A + ((size_t)(m0 + crow) * K) * 2 + cbyte;
  const char* Ab1 = Ab0 + (size_t)64 * K * 2;
  const char* Bb0 = (const char*)Bt + ((size_t)(n0 + crow) * K) * 2 + cbyte;
  const char* Bb1 = Bb0 + (size_t)64 * K * 2;
  const unsigned lA = LDS_OFF(&Asm[0][0]) + (unsigned)(crow * 64 + cbyte);
  const unsigned lB = LDS_OFF(&Bsm[0][0]) + (unsigned)(crow * 64 + cbyte);

  f32x8 acc[4][2];
#pragma unroll
  for (int ii = 0; ii < 4; ++ii)
#pragma unroll
    for (int jj = 0; jj < 2; ++jj)
#pragma unroll
      for (int t = 0; t < 8; ++t) acc[ii][jj][t] = 0.0f;

  const int nk = K / 32;

  // prologue: stage k-slice 0 into buffer 0
  {
    async_copy_b128(lA,        Ab0);
    async_copy_b128(lA + 4096, Ab1);
    async_copy_b128(lB,        Bb0);
    async_copy_b128(lB + 4096, Bb1);
  }

  for (int i = 0; i < nk; ++i) {
    const int cur = i & 1;
    if (i + 1 < nk) {
      const unsigned lbo = (unsigned)((cur ^ 1) * 128 * 64);
      const size_t go = (size_t)(i + 1) * 64;   // 32 bf16 = 64 bytes per k-step
      async_copy_b128(lA + lbo,        Ab0 + go);
      async_copy_b128(lA + lbo + 4096, Ab1 + go);
      async_copy_b128(lB + lbo,        Bb0 + go);
      async_copy_b128(lB + lbo + 4096, Bb1 + go);
      wait_async_le4();   // in-order async: current slice (issued earlier) done
    } else {
      wait_async_0();
    }
    __syncthreads();      // tile visible across all 8 waves

    bf16x16 af[4], bfr[2];
#pragma unroll
    for (int ii = 0; ii < 4; ++ii)
      af[ii] = load_frag16(&Asm[cur][(wm * 64 + ii * 16) * 32], 32, lane);
#pragma unroll
    for (int jj = 0; jj < 2; ++jj)
      bfr[jj] = load_frag16(&Bsm[cur][(wn * 32 + jj * 16) * 32], 32, lane);
#pragma unroll
    for (int ii = 0; ii < 4; ++ii)
#pragma unroll
      for (int jj = 0; jj < 2; ++jj)
        acc[ii][jj] = wmma_bf16(af[ii], bfr[jj], acc[ii][jj]);

    __syncthreads();      // everyone done reading before next overwrite
  }

  const int cl = lane & 15, ch = lane >> 4;
#pragma unroll
  for (int ii = 0; ii < 4; ++ii)
#pragma unroll
    for (int jj = 0; jj < 2; ++jj)
#pragma unroll
      for (int t = 0; t < 8; ++t) {
        int row = m0 + wm * 64 + ii * 16 + t + 8 * ch;
        int col = n0 + wn * 32 + jj * 16 + cl;
        C[(size_t)row * N + col] = acc[ii][jj][t] + bias[col];
      }
}

// ---------------------------------------------------------------------------
// Fused RMSNorm (over full DIM row) + RoPE + head-major packing of q,k
// One block per token. qkv fp32 [S][3*DIM]; outputs bf16 [H][S][D].
// ---------------------------------------------------------------------------
__global__ __launch_bounds__(256) void norm_rope_pack(const float* __restrict__ qkv,
                                                      const float* __restrict__ cosT,
                                                      const float* __restrict__ sinT,
                                                      const float* __restrict__ wq,
                                                      const float* __restrict__ wk,
                                                      bf16* __restrict__ qhm,
                                                      bf16* __restrict__ khm) {
  const int s = blockIdx.x, tid = threadIdx.x;
  const float* row = qkv + (size_t)s * N3_;

  float sq = 0.f, sk = 0.f;
  for (int c = tid; c < DIM_; c += 256) {
    float qv = row[c];        sq += qv * qv;
    float kv = row[DIM_ + c]; sk += kv * kv;
  }
  __shared__ float rq_[256], rk_[256];
  rq_[tid] = sq; rk_[tid] = sk;
  __syncthreads();
  for (int off = 128; off > 0; off >>= 1) {
    if (tid < off) { rq_[tid] += rq_[tid + off]; rk_[tid] += rk_[tid + off]; }
    __syncthreads();
  }
  const float rq = rsqrtf(rq_[0] / (float)DIM_ + 1e-6f);
  const float rk = rsqrtf(rk_[0] / (float)DIM_ + 1e-6f);

  for (int p = tid; p < DIM_ / 2; p += 256) {
    int c0 = 2 * p, c1 = c0 + 1;
    int h = c0 >> 7, d0 = c0 & 127;
    float cv = cosT[s * D_ + d0];
    float sv = sinT[s * D_ + d0 + 1];
    float qe = row[c0] * rq * wq[c0],        qo = row[c1] * rq * wq[c1];
    float ke = row[DIM_ + c0] * rk * wk[c0], ko = row[DIM_ + c1] * rk * wk[c1];
    size_t base = ((size_t)h * S_ + s) * D_ + d0;
    qhm[base]     = (bf16)(qe * cv - qo * sv);
    qhm[base + 1] = (bf16)(qe * sv + qo * cv);
    khm[base]     = (bf16)(ke * cv - ko * sv);
    khm[base + 1] = (bf16)(ke * sv + ko * cv);
  }
}

// ---------------------------------------------------------------------------
// Pack V transposed: qkv fp32 v-part -> vt bf16 [H][D][S] (keys contiguous)
// ---------------------------------------------------------------------------
__global__ void pack_v(const float* __restrict__ qkv, bf16* __restrict__ vt) {
  const int h = blockIdx.x;
  const int d0 = blockIdx.y * 32;
  const int s0 = blockIdx.z * 32;
  __shared__ float tile[32][33];
  const int tx = threadIdx.x, ty = threadIdx.y;
#pragma unroll
  for (int i = 0; i < 4; ++i) {
    int s = s0 + ty + i * 8, d = d0 + tx;
    tile[ty + i * 8][tx] = qkv[(size_t)s * N3_ + 2 * DIM_ + h * D_ + d];
  }
  __syncthreads();
#pragma unroll
  for (int i = 0; i < 4; ++i) {
    int d = d0 + ty + i * 8, s = s0 + tx;
    vt[((size_t)h * D_ + d) * S_ + s] = (bf16)tile[tx][ty + i * 8];
  }
}

// ---------------------------------------------------------------------------
// Flash attention: grid (S/128, H); 8 waves/block; each wave owns 16 queries.
// Streams keys in chunks of 32: two 16x16 score WMMAs, online softmax with
// 16-lane shuffle reductions, P transposed to A-layout through wave-private
// LDS, then 8 PV WMMAs accumulating the 16x128 output tile.
// ---------------------------------------------------------------------------
__global__ __launch_bounds__(256) void flash_attn(const bf16* __restrict__ qhm,
                                                  const bf16* __restrict__ khm,
                                                  const bf16* __restrict__ vt,
                                                  bf16* __restrict__ aout) {
  const int h = blockIdx.y;
  const int tid = threadIdx.x, lane = tid & 31, wave = tid >> 5;
  const int half = lane >> 4, cl = lane & 15;
  const int q0 = blockIdx.x * 128 + wave * 16;

  const bf16* qh = qhm + (size_t)h * S_ * D_;
  const bf16* kh = khm + (size_t)h * S_ * D_;
  const bf16* vh = vt  + (size_t)h * D_ * S_;

  bf16x16 qf[4];
#pragma unroll
  for (int kk = 0; kk < 4; ++kk)
    qf[kk] = load_frag16(qh + (size_t)q0 * D_ + kk * 32, D_, lane);

  f32x8 o[8];
  float mr[8], lr[8];
#pragma unroll
  for (int n = 0; n < 8; ++n)
#pragma unroll
    for (int t = 0; t < 8; ++t) o[n][t] = 0.0f;
#pragma unroll
  for (int t = 0; t < 8; ++t) { mr[t] = -1e30f; lr[t] = 0.0f; }

  __shared__ __align__(16) bf16 pbuf[8][16 * 32];  // 16x32 P tile per wave
  bf16* pb = &pbuf[wave][0];

  for (int j0 = 0; j0 < S_; j0 += 32) {
    f32x8 s0, s1;
#pragma unroll
    for (int t = 0; t < 8; ++t) { s0[t] = 0.0f; s1[t] = 0.0f; }

#pragma unroll
    for (int kk = 0; kk < 4; ++kk) {
      bf16x16 k0f = load_frag16(kh + (size_t)j0 * D_ + kk * 32, D_, lane);
      bf16x16 k1f = load_frag16(kh + (size_t)(j0 + 16) * D_ + kk * 32, D_, lane);
      s0 = wmma_bf16(qf[kk], k0f, s0);
      s1 = wmma_bf16(qf[kk], k1f, s1);
    }

#pragma unroll
    for (int t = 0; t < 8; ++t) {
      float a = s0[t] * SCALE_;
      float b = s1[t] * SCALE_;
      float mx = fmaxf(a, b);
#pragma unroll
      for (int off = 1; off < 16; off <<= 1) mx = fmaxf(mx, __shfl_xor(mx, off, 32));
      float mnew = fmaxf(mr[t], mx);
      float ea = __expf(a - mnew);
      float eb = __expf(b - mnew);
      float corr = __expf(mr[t] - mnew);
      float ps = ea + eb;
#pragma unroll
      for (int off = 1; off < 16; off <<= 1) ps += __shfl_xor(ps, off, 32);
      lr[t] = lr[t] * corr + ps;
      mr[t] = mnew;
#pragma unroll
      for (int n = 0; n < 8; ++n) o[n][t] *= corr;
      int prow = t + 8 * half;
      pb[prow * 32 + cl]      = (bf16)ea;
      pb[prow * 32 + 16 + cl] = (bf16)eb;
    }
    __syncthreads();

    bf16x16 pf = load_frag16(pb, 32, lane);
#pragma unroll
    for (int n = 0; n < 8; ++n) {
      bf16x16 vf = load_frag16(vh + (size_t)(n * 16) * S_ + j0, S_, lane);
      o[n] = wmma_bf16(pf, vf, o[n]);
    }
    __syncthreads();
  }

#pragma unroll
  for (int t = 0; t < 8; ++t) {
    float inv = 1.0f / lr[t];
    int row = q0 + t + 8 * half;
#pragma unroll
    for (int n = 0; n < 8; ++n)
      aout[(size_t)row * DIM_ + h * D_ + n * 16 + cl] = (bf16)(o[n][t] * inv);
  }
}

// ---------------------------------------------------------------------------
// Host launcher
// ---------------------------------------------------------------------------
extern "C" void kernel_launch(void* const* d_in, const int* in_sizes, int n_in,
                              void* d_out, int out_size, void* d_ws, size_t ws_size,
                              hipStream_t stream) {
  (void)in_sizes; (void)n_in; (void)out_size; (void)ws_size;

  const float* hs   = (const float*)d_in[0];  // (S, DIM)
  const float* cosT = (const float*)d_in[1];  // (S, 1, D)
  const float* sinT = (const float*)d_in[2];  // (S, 1, D)
  const float* wqkv = (const float*)d_in[3];  // (DIM, 3*DIM)
  const float* bqkv = (const float*)d_in[4];  // (3*DIM)
  const float* wqn  = (const float*)d_in[5];  // (DIM)
  const float* wkn  = (const float*)d_in[6];  // (DIM)
  const float* wout = (const float*)d_in[7];  // (DIM, DIM)
  const float* bout = (const float*)d_in[8];  // (DIM)
  float* out = (float*)d_out;                 // (S, DIM) fp32

  char* ws = (char*)d_ws;
  size_t off = 0;
  auto carve = [&](size_t bytes) -> void* {
    void* p = ws + off;
    off += (bytes + 255) & ~(size_t)255;
    return p;
  };
  bf16*  hsb    = (bf16*)carve((size_t)S_ * DIM_ * 2);        // bf16 activations
  bf16*  wqkvT  = (bf16*)carve((size_t)N3_ * DIM_ * 2);       // [3DIM][DIM]
  bf16*  woutT  = (bf16*)carve((size_t)DIM_ * DIM_ * 2);      // [DIM][DIM]
  float* qkvf   = (float*)carve((size_t)S_ * N3_ * 4);        // fp32 qkv
  bf16*  qhm    = (bf16*)carve((size_t)H_ * S_ * D_ * 2);     // [H][S][D]
  bf16*  khm    = (bf16*)carve((size_t)H_ * S_ * D_ * 2);     // [H][S][D]
  bf16*  vtb    = (bf16*)carve((size_t)H_ * D_ * S_ * 2);     // [H][D][S]
  bf16*  aob    = (bf16*)carve((size_t)S_ * DIM_ * 2);        // attn out bf16

  // 1) weight transpose+convert: wqkv [DIM][3DIM] -> [3DIM][DIM] bf16
  transpose_convert<<<dim3(N3_ / 32, DIM_ / 32), dim3(32, 8), 0, stream>>>(
      wqkv, wqkvT, DIM_, N3_);
  // 2) w_out [DIM][DIM] -> [DIM][DIM] bf16 transposed
  transpose_convert<<<dim3(DIM_ / 32, DIM_ / 32), dim3(32, 8), 0, stream>>>(
      wout, woutT, DIM_, DIM_);
  // 3) hidden -> bf16
  {
    int n = S_ * DIM_;
    convert_bf16<<<(n + 255) / 256, 256, 0, stream>>>(hs, hsb, n);
  }
  // 4) QKV GEMM: [S][3DIM] fp32
  gemm_bf16<<<dim3(N3_ / 128, S_ / 128), 256, 0, stream>>>(
      hsb, wqkvT, bqkv, qkvf, S_, N3_, DIM_);
  // 5) RMSNorm + RoPE + pack q,k head-major
  norm_rope_pack<<<S_, 256, 0, stream>>>(qkvf, cosT, sinT, wqn, wkn, qhm, khm);
  // 6) V transpose pack [H][D][S]
  pack_v<<<dim3(H_, D_ / 32, S_ / 32), dim3(32, 8), 0, stream>>>(qkvf, vtb);
  // 7) Flash attention -> bf16 [S][DIM]
  flash_attn<<<dim3(S_ / 128, H_), 256, 0, stream>>>(qhm, khm, vtb, aob);
  // 8) Output projection -> fp32 d_out
  gemm_bf16<<<dim3(DIM_ / 128, S_ / 128), 256, 0, stream>>>(
      aob, woutT, bout, out, S_, DIM_, DIM_);
}
